// Encoder_22986664968327
// MI455X (gfx1250) — matrix operand order repaired and checked by reading, not compiled, not run on hardware
//
#include <hip/hip_runtime.h>
#include <hip/hip_bf16.h>

typedef __attribute__((ext_vector_type(16))) __bf16 v16bf;
typedef __attribute__((ext_vector_type(8)))  __bf16 v8bf;
typedef __attribute__((ext_vector_type(8)))  float  v8f;

// ---------------- graph prep ----------------

__global__ __launch_bounds__(256) void k_fill_ones(float* __restrict__ deg, int n) {
  int i = blockIdx.x * blockDim.x + threadIdx.x;
  if (i < n) deg[i] = 1.0f;            // self-loop contributes 1 to every degree
}

__global__ __launch_bounds__(256) void k_degree(const int* __restrict__ dst,
                                                float* __restrict__ deg, int e) {
  int i = blockIdx.x * blockDim.x + threadIdx.x;
  if (i < e) atomicAdd(&deg[dst[i]], 1.0f);
}

__global__ __launch_bounds__(256) void k_rsqrt_inplace(float* __restrict__ d, int n) {
  int i = blockIdx.x * blockDim.x + threadIdx.x;
  if (i < n) d[i] = rsqrtf(d[i]);      // deg >= 1 always (self-loops)
}

// ---------------- bf16 WMMA GEMM:  Y[n_rows x N_OUT] = X[n_rows x K_DIM] @ W[K_DIM x N_OUT] ----------------
// 256 threads = 8 waves; block computes a 64-row tile x full N_OUT.
// wave w: row-subtile rt = w&3 (16 rows), column half ch = w>>2.
// Tail handling: only the last block is partial -> block-uniform scalar branch,
// full blocks take a guard-free path (straight ds_load_b128 + v_wmma + global_store).

template<int K_DIM, int N_OUT, bool RELU_IN>
__global__ __launch_bounds__(256) void gemm_bf16_wmma(
    const float* __restrict__ X, const float* __restrict__ W,
    float* __restrict__ Y, int n_rows) {
  constexpr int M_TILE = 64;
  constexpr int LDK = K_DIM + 8;            // pad 16B to skew LDS banks
  constexpr int NT = N_OUT / 16;
  constexpr int NT_PER_WAVE = NT / 2;

  __shared__ __bf16 sA[M_TILE * LDK];       // A tile, row-major [row][k]
  __shared__ __bf16 sB[N_OUT * LDK];        // W transposed: sB[n][k] = W[k][n]

  const int tid  = threadIdx.x;
  const int lane = tid & 31;
  const int wave = tid >> 5;
  const int rowBase = blockIdx.x * M_TILE;
  const bool fullTile = (rowBase + M_TILE) <= n_rows;   // block-uniform

  // stage weights (transposed) as bf16
  for (int idx = tid; idx < K_DIM * N_OUT; idx += 256) {
    const int k  = idx / N_OUT;
    const int nn = idx - k * N_OUT;
    sB[nn * LDK + k] = (__bf16)W[idx];
  }
  // stage activation tile as bf16 (fused ReLU); zero-pad OOB rows only in tail block
  if (fullTile) {
    for (int idx = tid; idx < M_TILE * K_DIM; idx += 256) {
      const int r = idx / K_DIM;
      const int k = idx - r * K_DIM;
      float v = X[(size_t)(rowBase + r) * K_DIM + k];
      if (RELU_IN) v = fmaxf(v, 0.0f);
      sA[r * LDK + k] = (__bf16)v;
    }
  } else {
    for (int idx = tid; idx < M_TILE * K_DIM; idx += 256) {
      const int r = idx / K_DIM;
      const int k = idx - r * K_DIM;
      const int row = rowBase + r;
      float v = (row < n_rows) ? X[(size_t)row * K_DIM + k] : 0.0f;
      if (RELU_IN) v = fmaxf(v, 0.0f);
      sA[r * LDK + k] = (__bf16)v;
    }
  }
  __syncthreads();

  const int rt  = wave & 3;
  const int ch  = wave >> 2;
  const int l15 = lane & 15;
  const int mrow = rt * 16 + l15;
  // ISA 7.12.2 16-bit A 16x32: lanes 0-15 hold K[0..7],K[16..23]; lanes 16-31 hold K[8..15],K[24..31]
  const int khA = (lane >= 16) ? 8 : 0;
  // B 32x16: lanes 0-15 hold K[0..7],K[16..23] of col N=lane; lanes 16-31 hold K[8..15],K[24..31] of col N=lane-16
  const int khB = (lane >= 16) ? 8 : 0;

  for (int t = 0; t < NT_PER_WAVE; ++t) {
    const int nt   = ch * NT_PER_WAVE + t;
    const int ncol = nt * 16 + l15;
    v8f acc = {};
#pragma unroll
    for (int kk = 0; kk < K_DIM; kk += 32) {
      const v8bf a0 = *(const v8bf*)&sA[mrow * LDK + kk + khA];
      const v8bf a1 = *(const v8bf*)&sA[mrow * LDK + kk + 16 + khA];
      const v8bf b0 = *(const v8bf*)&sB[ncol * LDK + kk + khB];
      const v8bf b1 = *(const v8bf*)&sB[ncol * LDK + kk + 16 + khB];
      v16bf a, b;
#pragma unroll
      for (int i = 0; i < 8; ++i) { a[i] = a0[i]; a[8 + i] = a1[i];
                                    b[i] = b0[i]; b[8 + i] = b1[i]; }
      acc = __builtin_amdgcn_wmma_f32_16x16x32_bf16(
          /*neg_a=*/false, a, /*neg_b=*/false, b,
          /*c_mod=*/(short)0, acc, /*reuse_a=*/false, /*reuse_b=*/false);
    }
    // D layout: VGPR r -> M = r (lanes 0-15) / 8+r (lanes 16-31), N = lane&15
    const int rbase = rowBase + rt * 16 + ((lane < 16) ? 0 : 8);
    float* __restrict__ yp = Y + (size_t)rbase * N_OUT + ncol;   // one 64-bit base per tile
    if (fullTile) {
#pragma unroll
      for (int r = 0; r < 8; ++r) yp[r * N_OUT] = acc[r];        // const imm offsets
    } else {
#pragma unroll
      for (int r = 0; r < 8; ++r) {
        if (rbase + r < n_rows) yp[r * N_OUT] = acc[r];
      }
    }
  }
}

// ---------------- GCN scatter:  out[i] = b + dinv[i]^2 * xw[i]  (self-loop), then edge atomics ----------------

template<int D>
__global__ __launch_bounds__(256) void k_self_bias_init(
    const float* __restrict__ xw, const float* __restrict__ dinv,
    const float* __restrict__ bias, float* __restrict__ out, int n) {
  constexpr int C4 = D / 4;
  const int idx = blockIdx.x * blockDim.x + threadIdx.x;
  if (idx >= n * C4) return;
  const int i = idx / C4;
  const int c = idx - i * C4;
  const float di = dinv[i];
  const float s = di * di;
  const float4 xv = ((const float4*)xw)[(size_t)i * C4 + c];
  const float4 bv = ((const float4*)bias)[c];
  float4 o;
  o.x = bv.x + xv.x * s; o.y = bv.y + xv.y * s;
  o.z = bv.z + xv.z * s; o.w = bv.w + xv.w * s;
  ((float4*)out)[(size_t)i * C4 + c] = o;
}

template<int D>
__global__ __launch_bounds__(256) void k_edge_scatter(
    const int* __restrict__ src, const int* __restrict__ dst,
    const float* __restrict__ xw, const float* __restrict__ dinv,
    float* __restrict__ out, int e) {
  constexpr int C4 = D / 4;
  const long long idx = (long long)blockIdx.x * blockDim.x + threadIdx.x;
  if (idx >= (long long)e * C4) return;
  const int ed = (int)(idx / C4);
  const int c  = (int)(idx - (long long)ed * C4);
  const int s = src[ed];
  const int d = dst[ed];
  const float w = dinv[s] * dinv[d];
  const float4 v = ((const float4*)xw)[(size_t)s * C4 + c];
  float* o = out + (size_t)d * D + (size_t)c * 4;
  atomicAdd(o + 0, v.x * w);
  atomicAdd(o + 1, v.y * w);
  atomicAdd(o + 2, v.z * w);
  atomicAdd(o + 3, v.w * w);
}

// ---------------- driver ----------------

extern "C" void kernel_launch(void* const* d_in, const int* in_sizes, int n_in,
                              void* d_out, int out_size, void* d_ws, size_t ws_size,
                              hipStream_t stream) {
  const float* x   = (const float*)d_in[0];
  const int*   ei  = (const int*)d_in[1];
  const float* W1  = (const float*)d_in[2];
  const float* b1  = (const float*)d_in[3];
  const float* W2  = (const float*)d_in[4];
  const float* b2  = (const float*)d_in[5];
  const float* Wmu = (const float*)d_in[6];
  const float* bmu = (const float*)d_in[7];
  const float* Wlv = (const float*)d_in[8];
  const float* blv = (const float*)d_in[9];

  const int n = in_sizes[0] / 64;     // N_NODES (IN_DIM = 64)
  const int E = in_sizes[1] / 2;      // edge_index is [2, E]
  const int* src = ei;
  const int* dst = ei + E;

  // workspace: dinv | bufA (N x 128) | bufB (N x 128)   (~103 MB)
  float* dinv = (float*)d_ws;
  const size_t nAl = ((size_t)n + 3) & ~(size_t)3;   // keep 16B alignment
  float* bufA = dinv + nAl;
  float* bufB = bufA + (size_t)n * 128;
  float* mu_out = (float*)d_out;
  float* lv_out = mu_out + (size_t)n * 64;

  const int T = 256;
  auto cd = [](long long a, int b) { return (int)((a + (long long)b - 1) / b); };

  // degree / normalization
  k_fill_ones<<<cd(n, T), T, 0, stream>>>(dinv, n);
  k_degree<<<cd(E, T), T, 0, stream>>>(dst, dinv, E);
  k_rsqrt_inplace<<<cd(n, T), T, 0, stream>>>(dinv, n);

  const int gB = cd(n, 64);

  // layer 1: h1_pre = scatter(x @ W1) + b1   (relu deferred into next GEMM load)
  gemm_bf16_wmma<64, 128, false><<<gB, T, 0, stream>>>(x, W1, bufA, n);
  k_self_bias_init<128><<<cd((long long)n * 32, T), T, 0, stream>>>(bufA, dinv, b1, bufB, n);
  k_edge_scatter<128><<<cd((long long)E * 32, T), T, 0, stream>>>(src, dst, bufA, dinv, bufB, E);

  // layer 2: h2_pre = scatter(relu(h1_pre) @ W2) + b2
  gemm_bf16_wmma<128, 128, true><<<gB, T, 0, stream>>>(bufB, W2, bufA, n);
  k_self_bias_init<128><<<cd((long long)n * 32, T), T, 0, stream>>>(bufA, dinv, b2, bufB, n);
  k_edge_scatter<128><<<cd((long long)E * 32, T), T, 0, stream>>>(src, dst, bufA, dinv, bufB, E);

  // mu = scatter(relu(h2_pre) @ Wmu) + bmu
  gemm_bf16_wmma<128, 64, true><<<gB, T, 0, stream>>>(bufB, Wmu, bufA, n);
  k_self_bias_init<64><<<cd((long long)n * 16, T), T, 0, stream>>>(bufA, dinv, bmu, mu_out, n);
  k_edge_scatter<64><<<cd((long long)E * 16, T), T, 0, stream>>>(src, dst, bufA, dinv, mu_out, E);

  // logvar = scatter(relu(h2_pre) @ Wlv) + blv
  gemm_bf16_wmma<128, 64, true><<<gB, T, 0, stream>>>(bufB, Wlv, bufA, n);
  k_self_bias_init<64><<<cd((long long)n * 16, T), T, 0, stream>>>(bufA, dinv, blv, lv_out, n);
  k_edge_scatter<64><<<cd((long long)E * 16, T), T, 0, stream>>>(src, dst, bufA, dinv, lv_out, E);
}
